// AttnBlock_17961553232295
// MI455X (gfx1250) — matrix operand order repaired and checked by reading, not compiled, not run on hardware
//
#include <hip/hip_runtime.h>
#include <hip/hip_bf16.h>
#include <stdint.h>

typedef __attribute__((ext_vector_type(16))) __bf16 v16bf;
typedef __attribute__((ext_vector_type(8)))  float  v8f;

union Frag {
  v16bf v;
  uint4 q[2];
  unsigned short u[16];
};

__device__ __forceinline__ unsigned short f2bf(float x) {
  unsigned int u = __float_as_uint(x);
  u += 0x7FFFu + ((u >> 16) & 1u);
  return (unsigned short)(u >> 16);
}

__device__ __forceinline__ v8f zero8() {
  v8f z = {0.f,0.f,0.f,0.f,0.f,0.f,0.f,0.f};
  return z;
}

// A fragment: 16x32 bf16 row-major. Per ISA 7.12.2, lane(m,half) holds
// K in [8h,8h+8) then [16+8h,16+8h+8) -> two contiguous b128 loads.
__device__ __forceinline__ v16bf load_a_frag(const unsigned short* A, int lda) {
  int lane = threadIdx.x & 31;
  int m = lane & 15, half = lane >> 4;
  Frag f;
  f.q[0] = *(const uint4*)(A + m * lda + 8 * half);
  f.q[1] = *(const uint4*)(A + m * lda + 16 + 8 * half);
  return f.v;
}

// B fragment from TRANSPOSED B (Bt is NxK row-major): lane(n,half) holds
// Bt[n][8h .. 8h+7] then Bt[n][16+8h .. 16+8h+7] -> two contiguous b128 loads.
__device__ __forceinline__ v16bf load_bt_frag(const unsigned short* Bt, int ldbt) {
  int lane = threadIdx.x & 31;
  int n = lane & 15, half = lane >> 4;
  Frag f;
  f.q[0] = *(const uint4*)(Bt + n * ldbt + 8 * half);
  f.q[1] = *(const uint4*)(Bt + n * ldbt + 16 + 8 * half);
  return f.v;
}

__device__ __forceinline__ v8f wmma_bf16(v16bf a, v16bf b, v8f c) {
  return __builtin_amdgcn_wmma_f32_16x16x32_bf16(false, a, false, b, (short)0, c, false, false);
}

// ---------------- workspace layout (bytes) ----------------
static constexpr size_t WS_HN    = 0;
static constexpr size_t WS_WQKV  = WS_HN    + (size_t)1024*256*2;   // Wqkv^T [768][256]
static constexpr size_t WS_WOUT  = WS_WQKV  + (size_t)256*768*2;    // Wout^T [256][256]
static constexpr size_t WS_W1A   = WS_WOUT  + (size_t)256*256*2;    // W1a^T [64][256]
static constexpr size_t WS_W1B   = WS_W1A   + (size_t)256*64*2;     // W1b^T [64][256]
static constexpr size_t WS_W1C   = WS_W1B   + (size_t)256*64*2;     // W1c^T [64][64]
static constexpr size_t WS_W2    = WS_W1C   + (size_t)64*64*2;      // W2^T  [64][64]
static constexpr size_t WS_Q     = WS_W2    + (size_t)64*64*2;      // q  [b,h,i,d]
static constexpr size_t WS_K     = WS_Q     + (size_t)2*8*512*32*2; // k  [b,h,i,d]  (natural = Bt for scores)
static constexpr size_t WS_VT    = WS_K     + (size_t)2*8*512*32*2; // v^T [b,h,d,i] (Bt for probs@v)
static constexpr size_t WS_HATT  = WS_VT    + (size_t)2*8*512*32*2;
static constexpr size_t WS_HNEW  = WS_HATT  + (size_t)1024*256*2;
static constexpr size_t WS_AROW  = WS_HNEW  + (size_t)1024*256*2;
static constexpr size_t WS_BROW  = WS_AROW  + (size_t)1024*64*4;
static constexpr size_t WS_EBIAS = WS_BROW  + (size_t)1024*64*4;
static constexpr size_t WS_ENBF  = WS_EBIAS + (size_t)2*8*512*512*4;

// ---------------- kernels ----------------

// transpose-convert: src f32 [K][N] row-major -> dst bf16 [N][K] row-major
__global__ void k_cvt_t(const float* __restrict__ s, unsigned short* __restrict__ d,
                        int K, int N) {
  int idx = blockIdx.x * blockDim.x + threadIdx.x;
  if (idx < K * N) {
    int k = idx / N, n = idx - k * N;
    d[n * K + k] = f2bf(s[idx]);
  }
}

// LayerNorm of h rows (C=256), output bf16
__global__ void k_ln_h(const float* __restrict__ h, const float* __restrict__ g,
                       const float* __restrict__ b, unsigned short* __restrict__ hn) {
  int row = blockIdx.x;          // 0..1023
  int t = threadIdx.x;           // 0..255
  float x = h[row * 256 + t];
  float s = x, q = x * x;
#pragma unroll
  for (int o = 16; o > 0; o >>= 1) { s += __shfl_xor(s, o, 32); q += __shfl_xor(q, o, 32); }
  __shared__ float ws_[8], wq_[8];
  int w = t >> 5;
  if ((t & 31) == 0) { ws_[w] = s; wq_[w] = q; }
  __syncthreads();
  float ts = 0.f, tq = 0.f;
#pragma unroll
  for (int i = 0; i < 8; ++i) { ts += ws_[i]; tq += wq_[i]; }
  float mean = ts * (1.0f / 256.0f);
  float var  = tq * (1.0f / 256.0f) - mean * mean;
  float inv  = rsqrtf(var + 1e-5f);
  hn[row * 256 + t] = f2bf((x - mean) * inv * g[t] + b[t]);
}

// qkv = hn @ Wqkv + bqkv ; each wave computes 4 N-tiles reusing the A fragment
__global__ void k_gemm_qkv(const unsigned short* __restrict__ hn,
                           const unsigned short* __restrict__ wqT,
                           const float* __restrict__ bqkv,
                           unsigned short* __restrict__ qb,
                           unsigned short* __restrict__ kb,
                           unsigned short* __restrict__ vT) {
  int w = threadIdx.x >> 5;
  int tileM = blockIdx.x & 63;
  int ngrp  = (blockIdx.x >> 6) * 4 + w;   // 0..11 -> tiles 4*ngrp..4*ngrp+3
  v8f acc[4] = {zero8(), zero8(), zero8(), zero8()};
#pragma unroll
  for (int kc = 0; kc < 8; ++kc) {
    v16bf a = load_a_frag(hn + tileM * 16 * 256 + kc * 32, 256);
#pragma unroll
    for (int t = 0; t < 4; ++t) {
      v16bf bb = load_bt_frag(wqT + ((ngrp * 4 + t) * 16) * 256 + kc * 32, 256);
      acc[t] = wmma_bf16(a, bb, acc[t]);
    }
  }
  int lane = threadIdx.x & 31;
  int n = lane & 15, half = lane >> 4;
#pragma unroll
  for (int t = 0; t < 4; ++t) {
    int tileN = ngrp * 4 + t;
#pragma unroll
    for (int r = 0; r < 8; ++r) {
      int m = tileM * 16 + r + 8 * half;
      int col = tileN * 16 + n;
      float val = acc[t][r] + bqkv[col];
      int which = col >> 8;
      int c2 = col & 255;
      int hh = c2 >> 5, d = c2 & 31;
      int bidx = m >> 9, i = m & 511;
      if (which == 0)      qb[(((bidx * 8 + hh) * 512) + i) * 32 + d] = f2bf(val * 0.17677669529663687f);
      else if (which == 1) kb[(((bidx * 8 + hh) * 512) + i) * 32 + d] = f2bf(val);
      else                 vT[(((bidx * 8 + hh) * 32) + d) * 512 + i] = f2bf(val);
    }
  }
}

// en = ln(e); edge_bias = ln(en@Web + beb); fold pad mask; 4 rows/block, 64 thr/row
__global__ void k_edge_ln_bias(const float* __restrict__ e,
                               const float* __restrict__ mask,
                               const float* __restrict__ ne_g, const float* __restrict__ ne_b,
                               const float* __restrict__ Web,  const float* __restrict__ beb,
                               const float* __restrict__ eb_g, const float* __restrict__ eb_b,
                               unsigned short* __restrict__ enb,
                               float* __restrict__ ebias) {
  int t64  = threadIdx.x & 63;
  int rloc = threadIdx.x >> 6;                      // 0..3
  long long row = (long long)blockIdx.x * 4 + rloc; // < 524288
  int j = (int)(row & 511);
  int i = (int)((row >> 9) & 511);
  int b = (int)(row >> 18);
  long long base = row * 64;
  float x = e[base + t64];
  float s = x, q = x * x;
#pragma unroll
  for (int o = 16; o > 0; o >>= 1) { s += __shfl_xor(s, o, 32); q += __shfl_xor(q, o, 32); }
  __shared__ float ws_[8], wq_[8];
  __shared__ float s_en[4][64];
  __shared__ float s_eb[4][8];
  int w = threadIdx.x >> 5;
  if ((threadIdx.x & 31) == 0) { ws_[w] = s; wq_[w] = q; }
  __syncthreads();
  int w0 = rloc << 1;
  float ts = ws_[w0] + ws_[w0 + 1];
  float tq = wq_[w0] + wq_[w0 + 1];
  float mean = ts * (1.0f / 64.0f);
  float var  = tq * (1.0f / 64.0f) - mean * mean;
  float inv  = rsqrtf(var + 1e-5f);
  float env = (x - mean) * inv * ne_g[t64] + ne_b[t64];
  enb[base + t64] = f2bf(env);
  s_en[rloc][t64] = env;
  __syncthreads();
  if (t64 < 8) {
    float acc = beb[t64];
#pragma unroll
    for (int d = 0; d < 64; ++d) acc += s_en[rloc][d] * Web[d * 8 + t64];
    s_eb[rloc][t64] = acc;
  }
  __syncthreads();
  if (t64 < 8) {
    float m8 = 0.f;
#pragma unroll
    for (int k = 0; k < 8; ++k) m8 += s_eb[rloc][k];
    m8 *= 0.125f;
    float v8 = 0.f;
#pragma unroll
    for (int k = 0; k < 8; ++k) { float dd = s_eb[rloc][k] - m8; v8 += dd * dd; }
    v8 *= 0.125f;
    float bias = (s_eb[rloc][t64] - m8) * rsqrtf(v8 + 1e-5f) * eb_g[t64] + eb_b[t64];
    float pad_i = (1.0f - mask[b * 512 + i]) * -1.0e9f;
    float pad_j = (1.0f - mask[b * 512 + j]) * -1.0e9f;
    ebias[((long long)(b * 8 + t64) * 512 + i) * 512 + j] = bias + pad_i + pad_j;
  }
}

// attention for one (b, h, 16-row q tile); 8 waves; scores + softmax + probs@v
__global__ void k_attn(const unsigned short* __restrict__ qb,
                       const unsigned short* __restrict__ kb,
                       const unsigned short* __restrict__ vT,
                       const float* __restrict__ ebias,
                       unsigned short* __restrict__ hatt) {
  int it = blockIdx.x & 31;
  int hh = (blockIdx.x >> 5) & 7;
  int b  = blockIdx.x >> 8;
  int w  = threadIdx.x >> 5;
  int lane = threadIdx.x & 31;
  int n = lane & 15, half = lane >> 4;

  __shared__ unsigned short sprob[16][512];
  __shared__ float smax[8][16];
  __shared__ float ssum[8][16];
  __shared__ float pout[8][16][32];
  __shared__ float gs[16];

  const unsigned short* qbase = qb + (((b * 8 + hh) * 512) + it * 16) * 32;
  v16bf aq = load_a_frag(qbase, 32);

  const float* ebbase = ebias + ((long long)(b * 8 + hh) * 512 + it * 16) * 512;
  v8f sc[4];
#pragma unroll
  for (int t = 0; t < 4; ++t) {
    int jt = w * 4 + t;
    // k natural layout [key][d] == transposed-B for scores (K=dh=32)
    v16bf bk = load_bt_frag(kb + (((b * 8 + hh) * 512) + jt * 16) * 32, 32);
    sc[t] = wmma_bf16(aq, bk, zero8());
#pragma unroll
    for (int r = 0; r < 8; ++r) {
      int m = r + 8 * half;
      sc[t][r] += ebbase[(long long)m * 512 + jt * 16 + n];
    }
  }
  // row max over this wave's 4 tiles, then cross-lane (within half) and cross-wave
  float rmax[8];
#pragma unroll
  for (int r = 0; r < 8; ++r) {
    float mm = sc[0][r];
#pragma unroll
    for (int t = 1; t < 4; ++t) mm = fmaxf(mm, sc[t][r]);
    rmax[r] = mm;
  }
#pragma unroll
  for (int o = 1; o < 16; o <<= 1)
#pragma unroll
    for (int r = 0; r < 8; ++r) rmax[r] = fmaxf(rmax[r], __shfl_xor(rmax[r], o, 32));
  if ((lane & 15) == 0) {
#pragma unroll
    for (int r = 0; r < 8; ++r) smax[w][8 * half + r] = rmax[r];
  }
  __syncthreads();
  float gmax[8];
#pragma unroll
  for (int r = 0; r < 8; ++r) {
    float mm = -3.0e38f;
#pragma unroll
    for (int ww = 0; ww < 8; ++ww) mm = fmaxf(mm, smax[ww][8 * half + r]);
    gmax[r] = mm;
  }
  float rsum[8];
#pragma unroll
  for (int r = 0; r < 8; ++r) rsum[r] = 0.f;
#pragma unroll
  for (int t = 0; t < 4; ++t) {
    int jt = w * 4 + t;
#pragma unroll
    for (int r = 0; r < 8; ++r) {
      float p = __expf(sc[t][r] - gmax[r]);
      rsum[r] += p;
      sprob[8 * half + r][jt * 16 + n] = f2bf(p);
    }
  }
#pragma unroll
  for (int o = 1; o < 16; o <<= 1)
#pragma unroll
    for (int r = 0; r < 8; ++r) rsum[r] += __shfl_xor(rsum[r], o, 32);
  if ((lane & 15) == 0) {
#pragma unroll
    for (int r = 0; r < 8; ++r) ssum[w][8 * half + r] = rsum[r];
  }
  __syncthreads();
  if (threadIdx.x < 16) {
    float ss = 0.f;
#pragma unroll
    for (int ww = 0; ww < 8; ++ww) ss += ssum[ww][threadIdx.x];
    gs[threadIdx.x] = ss;
  }
  // probs @ v : wave w handles key chunks 2w, 2w+1 ; two 16-col output tiles (dh=32)
  v8f o0 = zero8(), o1 = zero8();
#pragma unroll
  for (int kk = 0; kk < 2; ++kk) {
    int kc = w * 2 + kk;
    v16bf ap  = load_a_frag(&sprob[0][0] + kc * 32, 512);
    v16bf bv0 = load_bt_frag(vT + (((b * 8 + hh) * 32) + 0)  * 512 + kc * 32, 512);
    v16bf bv1 = load_bt_frag(vT + (((b * 8 + hh) * 32) + 16) * 512 + kc * 32, 512);
    o0 = wmma_bf16(ap, bv0, o0);
    o1 = wmma_bf16(ap, bv1, o1);
  }
#pragma unroll
  for (int r = 0; r < 8; ++r) {
    int m = r + 8 * half;
    pout[w][m][n]      = o0[r];
    pout[w][m][16 + n] = o1[r];
  }
  __syncthreads();
  for (int eidx = threadIdx.x; eidx < 512; eidx += 256) {
    int m = eidx >> 5, d = eidx & 31;
    float ssv = 0.f;
#pragma unroll
    for (int ww = 0; ww < 8; ++ww) ssv += pout[ww][m][d];
    float val = ssv / gs[m];
    hatt[((b * 512 + it * 16 + m) * 256) + hh * 32 + d] = f2bf(val);
  }
}

// h_new = h + hatt @ Wout + bout ; wave w covers N-tiles 4w..4w+3 (whole row of 16)
__global__ void k_out_proj(const unsigned short* __restrict__ hatt,
                           const unsigned short* __restrict__ woutT,
                           const float* __restrict__ bout,
                           const float* __restrict__ h0,
                           float* __restrict__ hnew,
                           unsigned short* __restrict__ hnewb) {
  int w = threadIdx.x >> 5;
  int tileM = blockIdx.x;        // 0..63
  v8f acc[4] = {zero8(), zero8(), zero8(), zero8()};
#pragma unroll
  for (int kc = 0; kc < 8; ++kc) {
    v16bf a = load_a_frag(hatt + tileM * 16 * 256 + kc * 32, 256);
#pragma unroll
    for (int t = 0; t < 4; ++t) {
      v16bf bb = load_bt_frag(woutT + ((w * 4 + t) * 16) * 256 + kc * 32, 256);
      acc[t] = wmma_bf16(a, bb, acc[t]);
    }
  }
  int lane = threadIdx.x & 31;
  int n = lane & 15, half = lane >> 4;
#pragma unroll
  for (int t = 0; t < 4; ++t) {
    int tileN = w * 4 + t;
#pragma unroll
    for (int r = 0; r < 8; ++r) {
      int m = tileM * 16 + r + 8 * half;
      int col = tileN * 16 + n;
      float val = acc[t][r] + bout[col] + h0[m * 256 + col];
      hnew[m * 256 + col]  = val;
      hnewb[m * 256 + col] = f2bf(val);
    }
  }
}

// Arow = h_new @ W1a + b1 (sel 0) ; Brow = h_new @ W1b (sel 1); wave covers N=64
__global__ void k_rowfacts(const unsigned short* __restrict__ hnewb,
                           const unsigned short* __restrict__ w1aT,
                           const unsigned short* __restrict__ w1bT,
                           const float* __restrict__ b1,
                           float* __restrict__ arow,
                           float* __restrict__ brow) {
  int gw = blockIdx.x * 4 + (threadIdx.x >> 5);  // 0..127
  int sel   = gw >> 6;   // 0 or 1
  int tileM = gw & 63;
  const unsigned short* wT = sel ? w1bT : w1aT;
  v8f acc[4] = {zero8(), zero8(), zero8(), zero8()};
#pragma unroll
  for (int kc = 0; kc < 8; ++kc) {
    v16bf a = load_a_frag(hnewb + tileM * 16 * 256 + kc * 32, 256);
#pragma unroll
    for (int t = 0; t < 4; ++t) {
      v16bf bb = load_bt_frag(wT + (t * 16) * 256 + kc * 32, 256);
      acc[t] = wmma_bf16(a, bb, acc[t]);
    }
  }
  int lane = threadIdx.x & 31;
  int n = lane & 15, half = lane >> 4;
  float* outp = sel ? brow : arow;
#pragma unroll
  for (int t = 0; t < 4; ++t) {
#pragma unroll
    for (int r = 0; r < 8; ++r) {
      int m = tileM * 16 + r + 8 * half;
      int col = t * 16 + n;
      float val = acc[t][r];
      if (sel == 0) val += b1[col];
      outp[m * 64 + col] = val;
    }
  }
}

// e_new = e + ln(relu(Arow_i + Brow_j + en@W1c)) @ W2 + b2
__global__ void k_edge_update(const unsigned short* __restrict__ enb,
                              const unsigned short* __restrict__ w1cT,
                              const unsigned short* __restrict__ w2T,
                              const float* __restrict__ arow,
                              const float* __restrict__ brow,
                              const float* __restrict__ mlp_g,
                              const float* __restrict__ mlp_b,
                              const float* __restrict__ b2,
                              const float* __restrict__ e0,
                              float* __restrict__ eout) {
  int jt128 = blockIdx.x & 3;
  int i = (blockIdx.x >> 2) & 511;
  int b = blockIdx.x >> 11;
  int w = threadIdx.x >> 5;
  int lane = threadIdx.x & 31;
  int n = lane & 15, half = lane >> 4;
  int j0 = jt128 * 128 + w * 16;

  __shared__ unsigned short s_w1c[64 * 64];   // W1c^T [64][64]
  __shared__ unsigned short s_w2[64 * 64];    // W2^T  [64][64]
  __shared__ unsigned short s_r[8][16 * 64];

  long long rowbase = ((long long)b * 512 + i) * 512 + j0;

  // warm the residual tile for the epilogue (global_prefetch_b8)
  __builtin_prefetch(e0 + (rowbase + (lane & 15)) * 64 + 16 * half, 0, 1);

  for (int idx = threadIdx.x; idx < 4096; idx += 256) {
    s_w1c[idx] = w1cT[idx];
    s_w2[idx]  = w2T[idx];
  }
  __syncthreads();

  const unsigned short* abase = enb + rowbase * 64;

  v8f acc[4] = {zero8(), zero8(), zero8(), zero8()};
#pragma unroll
  for (int kc = 0; kc < 2; ++kc) {
    v16bf a = load_a_frag(abase + kc * 32, 64);
#pragma unroll
    for (int nt = 0; nt < 4; ++nt) {
      v16bf bb = load_bt_frag(s_w1c + (nt * 16) * 64 + kc * 32, 64);
      acc[nt] = wmma_bf16(a, bb, acc[nt]);
    }
  }
  const float* ar = arow + ((long long)b * 512 + i) * 64;
  float fr[4][8];
#pragma unroll
  for (int nt = 0; nt < 4; ++nt) {
    int cidx = nt * 16 + n;
    float av = ar[cidx];
#pragma unroll
    for (int r = 0; r < 8; ++r) {
      int m = r + 8 * half;
      int j = j0 + m;
      float val = acc[nt][r] + av + brow[((long long)b * 512 + j) * 64 + cidx];
      fr[nt][r] = fmaxf(val, 0.f);
    }
  }
  // LayerNorm per row (j) over 64 cols
  float s1[8], s2[8];
#pragma unroll
  for (int r = 0; r < 8; ++r) {
    s1[r] = fr[0][r] + fr[1][r] + fr[2][r] + fr[3][r];
    s2[r] = fr[0][r]*fr[0][r] + fr[1][r]*fr[1][r] + fr[2][r]*fr[2][r] + fr[3][r]*fr[3][r];
  }
#pragma unroll
  for (int o = 1; o < 16; o <<= 1)
#pragma unroll
    for (int r = 0; r < 8; ++r) { s1[r] += __shfl_xor(s1[r], o, 32); s2[r] += __shfl_xor(s2[r], o, 32); }
#pragma unroll
  for (int nt = 0; nt < 4; ++nt) {
    int cidx = nt * 16 + n;
    float g = mlp_g[cidx], bb = mlp_b[cidx];
#pragma unroll
    for (int r = 0; r < 8; ++r) {
      int m = r + 8 * half;
      float mean = s1[r] * (1.0f / 64.0f);
      float var  = s2[r] * (1.0f / 64.0f) - mean * mean;
      float inv  = rsqrtf(var + 1e-5f);
      s_r[w][m * 64 + cidx] = f2bf((fr[nt][r] - mean) * inv * g + bb);
    }
  }
  // second GEMM: same-wave LDS region (LDS ops are in-order per wave)
  v8f acc2[4] = {zero8(), zero8(), zero8(), zero8()};
#pragma unroll
  for (int kc = 0; kc < 2; ++kc) {
    v16bf a = load_a_frag(&s_r[w][0] + kc * 32, 64);
#pragma unroll
    for (int nt = 0; nt < 4; ++nt) {
      v16bf bb = load_bt_frag(s_w2 + (nt * 16) * 64 + kc * 32, 64);
      acc2[nt] = wmma_bf16(a, bb, acc2[nt]);
    }
  }
#pragma unroll
  for (int nt = 0; nt < 4; ++nt) {
    int cidx = nt * 16 + n;
    float bv = b2[cidx];
#pragma unroll
    for (int r = 0; r < 8; ++r) {
      int m = r + 8 * half;
      long long idx = (rowbase + m) * 64 + cidx;
      eout[idx] = e0[idx] + acc2[nt][r] + bv;
    }
  }
}

// ---------------- launch ----------------
extern "C" void kernel_launch(void* const* d_in, const int* in_sizes, int n_in,
                              void* d_out, int out_size, void* d_ws, size_t ws_size,
                              hipStream_t stream) {
  const float* h    = (const float*)d_in[0];
  const float* e    = (const float*)d_in[1];
  const float* mask = (const float*)d_in[2];
  const float* Wqkv = (const float*)d_in[3];
  const float* bqkv = (const float*)d_in[4];
  const float* Wout = (const float*)d_in[5];
  const float* bout = (const float*)d_in[6];
  const float* Web  = (const float*)d_in[7];
  const float* beb  = (const float*)d_in[8];
  const float* eb_g = (const float*)d_in[9];
  const float* eb_b = (const float*)d_in[10];
  const float* W1   = (const float*)d_in[11];
  const float* b1   = (const float*)d_in[12];
  const float* mlp_g= (const float*)d_in[13];
  const float* mlp_b= (const float*)d_in[14];
  const float* W2   = (const float*)d_in[15];
  const float* b2   = (const float*)d_in[16];
  const float* nh_g = (const float*)d_in[17];
  const float* nh_b = (const float*)d_in[18];
  const float* ne_g = (const float*)d_in[19];
  const float* ne_b = (const float*)d_in[20];
  (void)in_sizes; (void)n_in; (void)out_size; (void)ws_size;

  char* ws = (char*)d_ws;
  unsigned short* hn    = (unsigned short*)(ws + WS_HN);
  unsigned short* wqkvT = (unsigned short*)(ws + WS_WQKV);
  unsigned short* woutT = (unsigned short*)(ws + WS_WOUT);
  unsigned short* w1aT  = (unsigned short*)(ws + WS_W1A);
  unsigned short* w1bT  = (unsigned short*)(ws + WS_W1B);
  unsigned short* w1cT  = (unsigned short*)(ws + WS_W1C);
  unsigned short* w2T   = (unsigned short*)(ws + WS_W2);
  unsigned short* qb    = (unsigned short*)(ws + WS_Q);
  unsigned short* kb    = (unsigned short*)(ws + WS_K);
  unsigned short* vT    = (unsigned short*)(ws + WS_VT);
  unsigned short* hattb = (unsigned short*)(ws + WS_HATT);
  unsigned short* hnewb = (unsigned short*)(ws + WS_HNEW);
  float* arow  = (float*)(ws + WS_AROW);
  float* brow  = (float*)(ws + WS_BROW);
  float* ebias = (float*)(ws + WS_EBIAS);
  unsigned short* enb = (unsigned short*)(ws + WS_ENBF);

  float* hnew = (float*)d_out;
  float* eout = (float*)d_out + (size_t)2 * 512 * 256;

  // bf16 transposed weight copies
  k_cvt_t<<<(196608 + 255) / 256, 256, 0, stream>>>(Wqkv,        wqkvT, 256, 768);
  k_cvt_t<<<(65536  + 255) / 256, 256, 0, stream>>>(Wout,        woutT, 256, 256);
  k_cvt_t<<<(16384  + 255) / 256, 256, 0, stream>>>(W1,          w1aT,  256, 64);
  k_cvt_t<<<(16384  + 255) / 256, 256, 0, stream>>>(W1 + 16384,  w1bT,  256, 64);
  k_cvt_t<<<(4096   + 255) / 256, 256, 0, stream>>>(W1 + 32768,  w1cT,  64,  64);
  k_cvt_t<<<(4096   + 255) / 256, 256, 0, stream>>>(W2,          w2T,   64,  64);

  k_ln_h<<<1024, 256, 0, stream>>>(h, nh_g, nh_b, hn);
  k_gemm_qkv<<<192, 128, 0, stream>>>(hn, wqkvT, bqkv, qb, kb, vT);
  k_edge_ln_bias<<<131072, 256, 0, stream>>>(e, mask, ne_g, ne_b, Web, beb, eb_g, eb_b, enb, ebias);
  k_attn<<<512, 256, 0, stream>>>(qb, kb, vT, ebias, hattb);
  k_out_proj<<<64, 128, 0, stream>>>(hattb, woutT, bout, h, hnew, hnewb);
  k_rowfacts<<<32, 128, 0, stream>>>(hnewb, w1aT, w1bT, b1, arow, brow);
  k_edge_update<<<4096, 256, 0, stream>>>(enb, w1cT, w2T, arow, brow, mlp_g, mlp_b, b2, e, eout);
}
